// ConvLSTMCell_83270825935290
// MI455X (gfx1250) — matrix operand order repaired and checked by reading, not compiled, not run on hardware
//
#include <hip/hip_runtime.h>

// ---------------------------------------------------------------------------
// Complex ConvLSTM cell, fused implicit-GEMM on CDNA5 WMMA (bf16 -> f32 acc).
//
// GEMM view:  M = B*H*W spatial positions, N = 384 gate channels
//             (i,o,c~ gates x {real,imag} x 64), K = 2304 (256 in-ch x 9 taps).
// f-gate inputs exist in setup but are unused by the reference -> skipped.
//
// Pipeline:
//   pack_weights : fold complex algebra ([Wr|-Wi ; Wi|Wr]) into bf16 Wp[t][n][k]
//   pack_z       : NCHW f32 -> NHWC bf16 Zp[b][h][w][ch] (LDS tile transpose)
//   convlstm_fused: per (b,h) row block; halo staged via async global->LDS DMA,
//                   9x8 k-steps of 12 v_wmma each, LDS-resident gate epilogue.
// ---------------------------------------------------------------------------

#define USE_ASYNC_LDS 1   // gfx1250 GLOBAL_LOAD_ASYNC_TO_LDS_B128 staging path

typedef __attribute__((ext_vector_type(16))) __bf16 bf16x16;
typedef __attribute__((ext_vector_type(8)))  float  f32x8;
typedef __attribute__((ext_vector_type(4)))  float  f32x4;

union Frag { bf16x16 bf; f32x4 f4[2]; };   // 32 bytes = one WMMA A/B operand

#define BATCH   16
#define H_IMG   64
#define W_IMG   64
#define KIN     256          // zr(128) ++ zi(128)
#define NOUT    384          // 3 gates * 128
#define TAPS    9
#define AS_CHS  264          // 256 channels + 8 pad (bank-conflict-free b128)
#define AS_BYTES (3*66*AS_CHS*2)
#define GS_STRIDE 65         // 64 + 1 pad
#define GS_BYTES  (NOUT*GS_STRIDE*4)
#define SMEM_BYTES (AS_BYTES > GS_BYTES ? AS_BYTES : GS_BYTES)
#define HWSZ 4096            // 64*64
#define PLANE ((size_t)BATCH*128*HWSZ)   // 8388608, h_new/c_new plane size

__device__ __forceinline__ unsigned short f32_to_bf16_rne(float f) {
  unsigned u = __float_as_uint(f);
  u += 0x7FFFu + ((u >> 16) & 1u);
  return (unsigned short)(u >> 16);
}

__device__ __forceinline__ float sigmoidf(float v) {
  return 1.0f / (1.0f + __expf(-v));
}

// ---------------------------------------------------------------------------
// Pack weights: Wp[t][n][k] bf16, n = gate(i,o,c)*128 + (real?0:64)+o,
// k = 0..127 -> zr (Wr for real rows, Wi for imag rows),
// k = 128..255 -> zi (-Wi for real rows, Wr for imag rows).
// Source layout OIHW: W[o][kin][ky][kx] -> (o*128 + kin)*9 + t.
// Also packs bias[384] (f32).
// ---------------------------------------------------------------------------
__global__ __launch_bounds__(256) void pack_weights(
    const float* __restrict__ Wr_i, const float* __restrict__ Wi_i,
    const float* __restrict__ Wr_o, const float* __restrict__ Wi_o,
    const float* __restrict__ Wr_c, const float* __restrict__ Wi_c,
    const float* __restrict__ br_i, const float* __restrict__ bi_i,
    const float* __restrict__ br_o, const float* __restrict__ bi_o,
    const float* __restrict__ br_c, const float* __restrict__ bi_c,
    unsigned short* __restrict__ Wp, float* __restrict__ biasOut)
{
  const int idx = blockIdx.x * 256 + threadIdx.x;
  if (idx < TAPS * NOUT * 256) {
    const int k = idx & 255;
    const int n = (idx >> 8) % NOUT;
    const int t = idx / (NOUT * 256);
    const int g = n >> 7;           // 0=i 1=o 2=c~
    const int r = n & 127;
    const int o = r & 63;
    const bool realOut = (r < 64);
    const float* Wr = (g == 0) ? Wr_i : (g == 1) ? Wr_o : Wr_c;
    const float* Wi = (g == 0) ? Wi_i : (g == 1) ? Wi_o : Wi_c;
    const bool zrPart = (k < 128);
    const int kin = zrPart ? k : (k - 128);
    const int src = (o * 128 + kin) * 9 + t;
    float wv;
    if (realOut) wv = zrPart ?  Wr[src] : -Wi[src];
    else         wv = zrPart ?  Wi[src] :  Wr[src];
    Wp[idx] = f32_to_bf16_rne(wv);
  }
  if (idx < NOUT) {
    const int g = idx >> 7, r = idx & 127, o = r & 63;
    const float* br = (g == 0) ? br_i : (g == 1) ? br_o : br_c;
    const float* bi = (g == 0) ? bi_i : (g == 1) ? bi_o : bi_c;
    biasOut[idx] = (r < 64) ? br[o] : bi[o];
  }
}

// ---------------------------------------------------------------------------
// Pack activations: z = [xr | hr | xi | hi] (256 ch) from NCHW f32 to
// NHWC bf16 Zp[b][h][w][ch] so the GEMM kernel can DMA contiguous 512B rows.
// One block per (b,h); LDS tile transpose keeps both sides coalesced.
// ---------------------------------------------------------------------------
__global__ __launch_bounds__(256) void pack_z(
    const float* __restrict__ x, const float* __restrict__ hp,
    unsigned short* __restrict__ Zp)
{
  __shared__ unsigned short T[64 * AS_CHS];   // [w][ch], padded stride
  const int b = blockIdx.x >> 6;
  const int h = blockIdx.x & 63;
  const int w = threadIdx.x & 63;

  for (int c = threadIdx.x >> 6; c < 256; c += 4) {   // coalesced along w
    const float* src; int cs;
    if      (c <  64) { src = x;  cs = c;       }     // xr
    else if (c < 128) { src = hp; cs = c - 64;  }     // hr
    else if (c < 192) { src = x;  cs = c - 64;  }     // xi (x ch 64..127)
    else              { src = hp; cs = c - 128; }     // hi (hp ch 64..127)
    const float v = src[(size_t)(b * 128 + cs) * HWSZ + h * 64 + w];
    T[w * AS_CHS + c] = f32_to_bf16_rne(v);
  }
  __syncthreads();

  unsigned short* dst = Zp + (size_t)((b * 64 + h) * 64) * 256;
  for (int i = threadIdx.x; i < 64 * 32; i += 256) {  // 64 rows x 32 b128 chunks
    const int ww = i >> 5, c16 = i & 31;
    *(f32x4*)(dst + ww * 256 + c16 * 8) = *(const f32x4*)(T + ww * AS_CHS + c16 * 8);
  }
}

// ---------------------------------------------------------------------------
// Fused kernel: one block per (b,h) output row. 8 wave32.
//  1) stage 3x66x256 bf16 halo into LDS via async global->LDS DMA (ASYNCcnt)
//  2) implicit-GEMM: 9 taps x 8 k-chunks, 12 v_wmma_f32_16x16x32_bf16 / step
//  3) acc+bias -> LDS, then complex-LSTM elementwise epilogue -> d_out
// ---------------------------------------------------------------------------
__global__ __launch_bounds__(256) void convlstm_fused(
    const float* __restrict__ x, const float* __restrict__ cp,
    const unsigned short* __restrict__ Zp,
    const unsigned short* __restrict__ Wp, const float* __restrict__ bias,
    float* __restrict__ out)
{
  extern __shared__ char smem[];
  unsigned short* As = (unsigned short*)smem;

  const int tid = threadIdx.x;
  const int b = blockIdx.x >> 6;
  const int h = blockIdx.x & 63;
  const int lane   = tid & 31;
  const int wv     = tid >> 5;       // wave id 0..7
  const int nl     = lane & 15;      // M index (A) / N index (B,C)
  const int hiHalf = lane >> 4;

  // ---- stage halo: As[(ri*66 + wp)*264 + ch], wp=0..65 <-> w=-1..64.
  // One 512B row per wave-iteration: 32 lanes x 16B. SAME-padding rows -> 0.
  {
    const unsigned ldsBase = (unsigned)(size_t)As;    // low 32b = LDS offset
    for (int rr = wv; rr < 3 * 66; rr += 8) {
      const int ri = rr / 66;
      const int wp = rr % 66;
      const int hr = h - 1 + ri;
      const int ww = wp - 1;
      if ((unsigned)hr < 64u && (unsigned)ww < 64u) {
#if USE_ASYNC_LDS
        const unsigned dstOff = ldsBase + (unsigned)(rr * AS_CHS * 2) + lane * 16;
        const unsigned long long gaddr =
            (unsigned long long)(size_t)(Zp + (size_t)((b * 64 + hr) * 64 + ww) * 256)
            + (unsigned)(lane * 16);
        asm volatile("global_load_async_to_lds_b128 %0, %1, off"
                     :: "v"(dstOff), "v"(gaddr) : "memory");
#else
        const unsigned short* srow =
            Zp + (size_t)((b * 64 + hr) * 64 + ww) * 256 + lane * 8;
        *(f32x4*)(As + rr * AS_CHS + lane * 8) = *(const f32x4*)srow;
#endif
      } else {
        f32x4 z4; z4[0] = 0.f; z4[1] = 0.f; z4[2] = 0.f; z4[3] = 0.f;
        *(f32x4*)(As + rr * AS_CHS + lane * 8) = z4;
      }
    }
#if USE_ASYNC_LDS
    asm volatile("s_wait_asynccnt 0" ::: "memory");   // own DMAs done before barrier
#endif
  }
  __syncthreads();

  f32x8 acc[4][3];
  #pragma unroll
  for (int mt = 0; mt < 4; ++mt)
    #pragma unroll
    for (int j = 0; j < 3; ++j)
      #pragma unroll
      for (int r = 0; r < 8; ++r) acc[mt][j][r] = 0.0f;

  int nIdx[3];
  #pragma unroll
  for (int j = 0; j < 3; ++j) nIdx[j] = (wv + j * 8) * 16 + nl;

  for (int t = 0; t < TAPS; ++t) {
    const int ri = t / 3;
    const int dx = t % 3 - 1;
    int aBase[4];
    #pragma unroll
    for (int mt = 0; mt < 4; ++mt)
      aBase[mt] = (ri * 66 + (mt * 16 + nl + dx + 1)) * AS_CHS + hiHalf * 8;
    int bBase[3];
    #pragma unroll
    for (int j = 0; j < 3; ++j) {
      bBase[j] = (t * NOUT + nIdx[j]) * 256 + hiHalf * 8;
      if (t + 1 < TAPS)   // warm L2 path for next tap's weights
        __builtin_prefetch(Wp + bBase[j] + NOUT * 256, 0, 1);
    }

    for (int kc = 0; kc < 8; ++kc) {
      const int kOff = kc * 32;
      Frag A[4], Bm[3];
      #pragma unroll
      for (int mt = 0; mt < 4; ++mt) {           // A frag: K{c0..c0+7, 16+c0..}
        A[mt].f4[0] = *(const f32x4*)(As + aBase[mt] + kOff);
        A[mt].f4[1] = *(const f32x4*)(As + aBase[mt] + kOff + 16);
      }
      #pragma unroll
      for (int j = 0; j < 3; ++j) {              // B frag: K{c0..c0+7, 16+c0..} at col n
        Bm[j].f4[0] = *(const f32x4*)(Wp + bBase[j] + kOff);
        Bm[j].f4[1] = *(const f32x4*)(Wp + bBase[j] + kOff + 16);
      }
      #pragma unroll
      for (int mt = 0; mt < 4; ++mt)
        #pragma unroll
        for (int j = 0; j < 3; ++j)
          acc[mt][j] = __builtin_amdgcn_wmma_f32_16x16x32_bf16(
              false, A[mt].bf, false, Bm[j].bf, (short)0, acc[mt][j],
              false, false);
    }
  }

  __syncthreads();                     // done reading As; reuse LDS for gates

  float* Gs = (float*)smem;            // Gs[n][w], stride 65
  float bv[3];
  #pragma unroll
  for (int j = 0; j < 3; ++j) bv[j] = bias[nIdx[j]];

  #pragma unroll
  for (int mt = 0; mt < 4; ++mt)
    #pragma unroll
    for (int j = 0; j < 3; ++j)
      #pragma unroll
      for (int r = 0; r < 8; ++r) {    // C layout: lane holds N=nl, M=r+8*hiHalf
        const int w = mt * 16 + r + hiHalf * 8;
        Gs[nIdx[j] * GS_STRIDE + w] = acc[mt][j][r] + bv[j];
      }
  __syncthreads();

  // ---- complex LSTM elementwise epilogue (all gates local in LDS) ----
  for (int idx = tid; idx < 4096; idx += 256) {
    const int o = idx >> 6, w = idx & 63;
    const float i_r  = sigmoidf(Gs[(      o) * GS_STRIDE + w]);
    const float i_i  = sigmoidf(Gs[( 64 + o) * GS_STRIDE + w]);
    const float o_r  = sigmoidf(Gs[(128 + o) * GS_STRIDE + w]);
    const float o_i  = sigmoidf(Gs[(192 + o) * GS_STRIDE + w]);
    const float ct_r = tanhf   (Gs[(256 + o) * GS_STRIDE + w]);
    const float ct_i = tanhf   (Gs[(320 + o) * GS_STRIDE + w]);

    const size_t pr = (size_t)(b * 128 +      o) * HWSZ + h * 64 + w;
    const size_t pi = (size_t)(b * 128 + 64 + o) * HWSZ + h * 64 + w;
    const float xr = x[pr], xi = x[pi];
    const float cr = cp[pr], ci = cp[pi];

    const float cnr = xr * cr - xi * ci + i_r * ct_r - i_i * ct_i;
    const float cni = xr * ci + xi * cr + i_r * ct_i + i_i * ct_r;
    const float tr = tanhf(cnr), ti = tanhf(cni);

    out[pr]         = o_r * tr - o_i * ti;   // h_new real
    out[pi]         = o_r * ti + o_i * tr;   // h_new imag
    out[PLANE + pr] = cnr;                   // c_new real
    out[PLANE + pi] = cni;                   // c_new imag
  }
}

// ---------------------------------------------------------------------------
extern "C" void kernel_launch(void* const* d_in, const int* in_sizes, int n_in,
                              void* d_out, int out_size, void* d_ws, size_t ws_size,
                              hipStream_t stream) {
  (void)in_sizes; (void)n_in; (void)out_size; (void)ws_size;

  const float* x    = (const float*)d_in[0];
  const float* hpv  = (const float*)d_in[1];
  const float* cpv  = (const float*)d_in[2];
  const float* Wr_i = (const float*)d_in[3];
  const float* Wi_i = (const float*)d_in[4];
  const float* br_i = (const float*)d_in[5];
  const float* bi_i = (const float*)d_in[6];
  // d_in[7..10] = f-gate tensors: unused by the reference computation
  const float* Wr_o = (const float*)d_in[11];
  const float* Wi_o = (const float*)d_in[12];
  const float* br_o = (const float*)d_in[13];
  const float* bi_o = (const float*)d_in[14];
  const float* Wr_c = (const float*)d_in[15];
  const float* Wi_c = (const float*)d_in[16];
  const float* br_c = (const float*)d_in[17];
  const float* bi_c = (const float*)d_in[18];

  // workspace layout
  unsigned short* Wp  = (unsigned short*)d_ws;                      // 1.77 MB
  float* bias = (float*)((char*)d_ws + (size_t)TAPS * NOUT * 256 * 2);
  unsigned short* Zp  = (unsigned short*)((char*)d_ws +
                        (size_t)TAPS * NOUT * 256 * 2 + NOUT * 4);  // 33.5 MB

  pack_weights<<<(TAPS * NOUT * 256 + 255) / 256, 256, 0, stream>>>(
      Wr_i, Wi_i, Wr_o, Wi_o, Wr_c, Wi_c,
      br_i, bi_i, br_o, bi_o, br_c, bi_c, Wp, bias);

  pack_z<<<BATCH * H_IMG, 256, 0, stream>>>(x, hpv, Zp);

  convlstm_fused<<<BATCH * H_IMG, 256, SMEM_BYTES, stream>>>(
      x, cpv, Zp, Wp, bias, (float*)d_out);
}
